// CTRNN_78821239816331
// MI455X (gfx1250) — compile-verified, matrix operand order checked by reading, not run on hardware
//
#include <hip/hip_runtime.h>

// ---------------------------------------------------------------------------
// CTRNN on MI455X (gfx1250, wave32).
// alpha = 1/TAU = 1.0  =>  u_{t+1} = drive + z_t @ H_w^T ; z = tanh(u)
//
// Per workgroup (512 threads = 16 waves): 64 batch rows. z_t kept in LDS
// (f16, row-major), ping-ponged between two 64KB buffers -> ONE barrier/step.
// Wave w = (M-tile w&3, N-quarter w>>2): 8 accumulator tiles (64 VGPRs).
// Inner loop: kc loop FORCED rolled (unroll(disable) -- letting clang unroll
// it explodes register pressure and spills acc/B to scratch), unrolled n loop
// with an explicit distance-1 carried B double-buffer: the load of fragment
// (kc,n+1) (or (kc+1,0)) is issued before the WMMA consuming (kc,n), so one
// fragment (2x global_load_b128) is always in flight per wave, x4 waves/SIMD.
// H_w pre-packed once (f32 -> f16, WMMA-B fragment order) into d_ws;
// L2-resident, streamed through WGP$ with 4-way intra-WGP wave reuse.
// ---------------------------------------------------------------------------

typedef _Float16 h16;
typedef __attribute__((ext_vector_type(16))) _Float16 v16h;
typedef __attribute__((ext_vector_type(8)))  _Float16 v8h;
typedef __attribute__((ext_vector_type(8)))  float    v8f;

#define HID    512
#define OUTD   12
#define TSTEPS 256
#define TILE_B 64          // batch rows per workgroup
#define KCH    16          // 512 / 32 K-chunks
#define NPW    8           // N-tiles per wave

// ---- gfx1250 transcendentals (TRANS pipe) ---------------------------------
__device__ __forceinline__ float fast_tanh(float x) {
#if __has_builtin(__builtin_amdgcn_tanhf)
  return __builtin_amdgcn_tanhf(x);
#else
  float r;
  asm volatile("v_tanh_f32 %0, %1\n\tv_nop\n\tv_nop" : "=v"(r) : "v"(x));
  return r;
#endif
}
__device__ __forceinline__ float fast_exp2(float x) {
#if __has_builtin(__builtin_amdgcn_exp2f)
  return __builtin_amdgcn_exp2f(x);
#else
  float r;
  asm volatile("v_exp_f32 %0, %1\n\tv_nop\n\tv_nop" : "=v"(r) : "v"(x));
  return r;
#endif
}
__device__ __forceinline__ float fast_rcp(float x) {
#if __has_builtin(__builtin_amdgcn_rcpf)
  return __builtin_amdgcn_rcpf(x);
#else
  float r;
  asm volatile("v_rcp_f32 %0, %1\n\tv_nop\n\tv_nop" : "=v"(r) : "v"(x));
  return r;
#endif
}
__device__ __forceinline__ float fast_sigmoid(float x) {
  return fast_rcp(1.0f + fast_exp2(-1.442695041f * x));
}

// K-index mapping inside a 32-wide chunk for 16-bit WMMA fragments:
//   element j of lane ln  <->  K = (ln>>4)*8 + (j<8 ? j : j+8)
__device__ __forceinline__ int kmap(int j, int g) {
  return g * 8 + (j < 8 ? j : j + 8);
}

// ---- pack H_w (f32 [512,512], row n = output unit) into f16 B-fragments ----
// Fragment f = kc*32 + ntile: 32 lanes x 16 halves, contiguous 1 KB each.
__global__ __launch_bounds__(256) void pack_H(const float* __restrict__ Hw,
                                              h16* __restrict__ Hp) {
  int lane = threadIdx.x & 31, sw = threadIdx.x >> 5;
  int frag = blockIdx.x * 8 + sw;          // 0..511
  int kc = frag >> 5, nt = frag & 31;
  int g = lane >> 4, col = lane & 15;
  int n = nt * 16 + col;                   // B column = output unit index
#pragma unroll
  for (int j = 0; j < 16; ++j) {
    int K = kc * 32 + kmap(j, g);
    Hp[(unsigned)frag * 512u + lane * 16 + j] = (h16)Hw[(unsigned)n * HID + K];
  }
}

// ---- pack O_w (f32 [12,512]) into f16 B-fragments, N padded 12 -> 16 ------
__global__ __launch_bounds__(256) void pack_O(const float* __restrict__ Ow,
                                              h16* __restrict__ Op) {
  int lane = threadIdx.x & 31, sw = threadIdx.x >> 5;
  int kc = blockIdx.x * 8 + sw;            // 0..15
  if (kc >= KCH) return;
  int g = lane >> 4, col = lane & 15;
#pragma unroll
  for (int j = 0; j < 16; ++j) {
    int K = kc * 32 + kmap(j, g);
    float val = (col < OUTD) ? Ow[(unsigned)col * HID + K] : 0.0f;
    Op[(unsigned)kc * 512u + lane * 16 + j] = (h16)val;
  }
}

// ---- A-fragment load from an LDS z buffer (row-major f16) -----------------
__device__ __forceinline__ v16h load_a(const h16* zbuf, int row, int kc, int g) {
  const h16* za = zbuf + (unsigned)row * HID + kc * 32 + g * 8;
  v8h lo = *(const v8h*)za;
  v8h hi = *(const v8h*)(za + 16);
  v16h a;
#pragma unroll
  for (int j = 0; j < 8; ++j) { a[j] = lo[j]; a[j + 8] = hi[j]; }
  return a;
}

// ---- main recurrent kernel (512 threads = 16 waves) -----------------------
__global__ __launch_bounds__(512) void ctrnn_main(const float* __restrict__ x,
                                                  const float* __restrict__ Iw,
                                                  const float* __restrict__ vb,
                                                  const float* __restrict__ mb,
                                                  const h16* __restrict__ Hp,
                                                  const h16* __restrict__ Op,
                                                  float* __restrict__ out) {
  extern __shared__ h16 Zls[];             // 2 x 64KB ping-pong z buffers

  const int tid  = threadIdx.x;
  const int lane = tid & 31;
  const int wave = tid >> 5;               // 16 waves
  const int mt   = wave & 3;               // M-tile (16 batch rows)
  const int nq   = wave >> 2;              // N-quarter (8 N-tiles each)
  const int g    = lane >> 4;
  const int col  = lane & 15;
  const int bb   = blockIdx.x * TILE_B;

  // per-wave B base: fragments (kc*32 + nq*8 + n), this lane's 32B slice
  const h16* Hq = Hp + (unsigned)nq * (NPW * 512u) + lane * 16;
  const h16* Oq = Op + lane * 16;

  h16* cur = Zls;                          // holds z_t
  h16* nxt = Zls + TILE_B * HID;           // receives z_{t+1}

  // zero z_0 in cur
  {
    int4* zi = (int4*)cur;
    for (int i = tid; i < (TILE_B * HID * 2) / 16; i += 512)
      zi[i] = int4{0, 0, 0, 0};
  }

  // per-lane drive constants in WMMA-C layout:
  //   C element (reg r, lane ln): M = r + 8*(ln>>4), N = ln&15
  float xr[8];
#pragma unroll
  for (int r = 0; r < 8; ++r) xr[r] = x[bb + mt * 16 + r + g * 8];
  float iw[NPW], vv[NPW];
#pragma unroll
  for (int n = 0; n < NPW; ++n) {
    int h = nq * (NPW * 16) + n * 16 + col;
    iw[n] = Iw[h];
    vv[n] = vb[h];
  }
  const float mo = (col < OUTD) ? mb[col] : 0.0f;

  __syncthreads();                         // z_0 visible to all waves

#pragma clang loop unroll(disable)
  for (int t = 0; t < TSTEPS; ++t) {
    // -------- u = drive + z_t @ H_w^T  (wave's 16x128 slice) --------
    v8f acc[NPW];
#pragma unroll
    for (int n = 0; n < NPW; ++n) {
#pragma unroll
      for (int r = 0; r < 8; ++r) acc[n][r] = fmaf(xr[r], iw[n], vv[n]);
    }

    const int arow = mt * 16 + col;

    // Distance-1 software pipeline: bcur holds fragment (kc,n); the load of
    // (kc,n+1) / (kc+1,0) is issued before the WMMA that consumes bcur.
    // At n==7 the preload is unconditional: for kc==15 it reads a harmless
    // dummy just past Hp (in Op for nq<2, beyond Op for nq>=2) and is never consumed.
    v16h bcur = *(const v16h*)(Hq);        // fragment (0,0)
#pragma clang loop unroll(disable)
    for (int kc = 0; kc < KCH; ++kc) {
      v16h a = load_a(cur, arow, kc, g);
      const h16* base = Hq + (unsigned)kc * (32u * 512u);
#pragma unroll
      for (int n = 0; n < NPW; ++n) {
        v16h bnext = (n + 1 < NPW)
                         ? *(const v16h*)(base + (unsigned)(n + 1) * 512u)
                         : *(const v16h*)(base + 32u * 512u);  // (kc+1, 0)
        acc[n] = __builtin_amdgcn_wmma_f32_16x16x32_f16(
            false, a, false, bcur, (short)0, acc[n], false, false);
        bcur = bnext;
      }
    }

    // -------- z_{t+1} = tanh(u) -> nxt (disjoint from cur: no barrier) ----
#pragma unroll
    for (int n = 0; n < NPW; ++n) {
      int hbase = nq * (NPW * 16) + n * 16 + col;
#pragma unroll
      for (int r = 0; r < 8; ++r) {
        int row = mt * 16 + r + g * 8;
        nxt[(unsigned)row * HID + hbase] = (h16)fast_tanh(acc[n][r]);
      }
    }

    __syncthreads();   // z_{t+1} complete; also fences reuse of 'cur' next step

    // -------- y_t = sigmoid(z_{t+1} @ O_w^T + m), waves 0..3 --------------
    // (pure reads of nxt; overlaps waves 4..15 starting step t+1)
    if (wave < 4) {
      const int omt = wave;
      v8f co;
#pragma unroll
      for (int r = 0; r < 8; ++r) co[r] = mo;
      const int orow = omt * 16 + col;
#pragma clang loop unroll(disable)
      for (int kc = 0; kc < KCH; ++kc) {
        v16h a = load_a(nxt, orow, kc, g);
        v16h b = *(const v16h*)(Oq + (unsigned)kc * 512u);
        co = __builtin_amdgcn_wmma_f32_16x16x32_f16(
            false, a, false, b, (short)0, co, false, false);
      }
      if (col < OUTD) {
#pragma unroll
        for (int r = 0; r < 8; ++r) {
          int b_ = bb + omt * 16 + r + g * 8;
          out[(unsigned)b_ * (TSTEPS * OUTD) + (unsigned)t * OUTD + col] =
              fast_sigmoid(co[r]);
        }
      }
    }

    // swap ping-pong buffers
    h16* tmp = cur; cur = nxt; nxt = tmp;
  }
}

// ---------------------------------------------------------------------------
extern "C" void kernel_launch(void* const* d_in, const int* in_sizes, int n_in,
                              void* d_out, int out_size, void* d_ws,
                              size_t ws_size, hipStream_t stream) {
  // setup_inputs order: x, T, I_w, H_w, O_w, v, m
  const float* x  = (const float*)d_in[0];
  const float* Iw = (const float*)d_in[2];
  const float* Hw = (const float*)d_in[3];
  const float* Ow = (const float*)d_in[4];
  const float* vb = (const float*)d_in[5];
  const float* mb = (const float*)d_in[6];

  h16* Hp = (h16*)d_ws;                 // 512*512 f16 = 512 KB
  h16* Op = Hp + (size_t)HID * HID;     // 16*512 f16  =  16 KB

  pack_H<<<64, 256, 0, stream>>>(Hw, Hp);
  pack_O<<<2, 256, 0, stream>>>(Ow, Op);

  const int B = in_sizes[0];            // 8192
  const size_t lds_bytes = 2u * TILE_B * HID * sizeof(h16);  // 128 KB
  ctrnn_main<<<B / TILE_B, 512, lds_bytes, stream>>>(x, Iw, vb, mb, Hp, Op,
                                                     (float*)d_out);
}
